// JBUStack_86414741996331
// MI455X (gfx1250) — compile-verified, matrix operand order checked by reading, not run on hardware
//
#include <hip/hip_runtime.h>
#include <hip/hip_bf16.h>

#define RAD 3
#define DIA 7
#define D2  49
#define KD  32
#define CC  384
#define BB  4
#define TS  16
#define PAD 22   // TS + 2*RAD

typedef __attribute__((ext_vector_type(16))) __bf16 v16bf;
typedef __attribute__((ext_vector_type(8)))  __bf16 v8bf;
typedef __attribute__((ext_vector_type(8)))  float  v8f;

__device__ __forceinline__ int refl(int t, int n) {
    if (t < 0) t = -t;
    if (t >= n) t = 2 * n - 2 - t;
    return t;
}
__device__ __forceinline__ int clampi(int v, int lo, int hi) {
    return v < lo ? lo : (v > hi ? hi : v);
}
__device__ __forceinline__ float gelu_exact(float x) {
    return 0.5f * x * (1.0f + erff(x * 0.70710678118654752f));
}
__device__ __forceinline__ float cubw(float x) {
    const float a = -0.75f;
    float ax = fabsf(x);
    if (ax <= 1.0f) return ((a + 2.0f) * ax - (a + 3.0f)) * ax * ax + 1.0f;
    if (ax < 2.0f)  return a * (((ax - 5.0f) * ax + 8.0f) * ax - 4.0f);
    return 0.0f;
}
// torch bicubic x2, align_corners=False: src coord = dst*0.5 - 0.25, edge clamp
__device__ __forceinline__ float bicubic2x(const float* __restrict__ src,
                                           int h, int w, int ry, int rx) {
    float sy = ry * 0.5f - 0.25f;
    float sx = rx * 0.5f - 0.25f;
    int iy = (int)floorf(sy), ix = (int)floorf(sx);
    float ty = sy - (float)iy, tx = sx - (float)ix;
    float wy[4] = { cubw(ty + 1.0f), cubw(ty), cubw(1.0f - ty), cubw(2.0f - ty) };
    float wx[4] = { cubw(tx + 1.0f), cubw(tx), cubw(1.0f - tx), cubw(2.0f - tx) };
    float acc = 0.0f;
    #pragma unroll
    for (int i = 0; i < 4; i++) {
        int yy = clampi(iy - 1 + i, 0, h - 1);
        float row = 0.0f;
        #pragma unroll
        for (int j = 0; j < 4; j++) {
            int xx = clampi(ix - 1 + j, 0, w - 1);
            row += wx[j] * src[yy * w + xx];
        }
        acc += wy[i] * row;
    }
    return acc;
}

// ---------------- adaptive avg pool of guidance (224 -> H) ----------------
__global__ void pool_kernel(const float* __restrict__ gsrc, float* __restrict__ g,
                            int H, int W, int f) {
    int idx = blockIdx.x * 256 + threadIdx.x;
    int n = BB * 3 * H * W;
    if (idx >= n) return;
    int x = idx % W;
    int y = (idx / W) % H;
    int c = (idx / (W * H)) % 3;
    int b = idx / (W * H * 3);
    const float* src = gsrc + ((size_t)(b * 3 + c) * 224 + (size_t)y * f) * 224 + (size_t)x * f;
    float s = 0.0f;
    for (int dy = 0; dy < f; dy++)
        for (int dx = 0; dx < f; dx++)
            s += src[dy * 224 + dx];
    g[idx] = s / (float)(f * f);
}

// ---------------- range projection MLP: 3 -> 32 (gelu) -> 32 ----------------
__global__ void proj_kernel(const float* __restrict__ g,
                            const float* __restrict__ w1, const float* __restrict__ b1,
                            const float* __restrict__ w2, const float* __restrict__ b2,
                            float* __restrict__ proj, int H, int W) {
    int idx = blockIdx.x * 256 + threadIdx.x;
    int n = BB * H * W;
    if (idx >= n) return;
    int x = idx % W;
    int y = (idx / W) % H;
    int b = idx / (W * H);
    float gi[3];
    #pragma unroll
    for (int c = 0; c < 3; c++) gi[c] = g[((size_t)(b * 3 + c) * H + y) * W + x];
    float h1[KD];
    #pragma unroll
    for (int o = 0; o < KD; o++) {
        float a = b1[o] + w1[o * 3] * gi[0] + w1[o * 3 + 1] * gi[1] + w1[o * 3 + 2] * gi[2];
        h1[o] = gelu_exact(a);
    }
    for (int o = 0; o < KD; o++) {
        float a = b2[o];
        const float* wr = w2 + o * KD;
        #pragma unroll
        for (int i2 = 0; i2 < KD; i2++) a += wr[i2] * h1[i2];
        proj[((size_t)(b * KD + o) * H + y) * W + x] = a;
    }
}

// ------- 49-offset range logits + softmax + spatial gaussian + normalize -------
__global__ void ck_kernel(const float* __restrict__ proj,
                          const float* __restrict__ rt, const float* __restrict__ ss,
                          float* __restrict__ ck, int H, int W) {
    __shared__ float sp[PAD * PAD * KD];
    int b = blockIdx.z;
    int x0 = blockIdx.x * TS, y0 = blockIdx.y * TS;
    int tid = threadIdx.y * TS + threadIdx.x;
    const int total = PAD * PAD * KD;
    for (int e = tid; e < total; e += 256) {
        int k = e & (KD - 1);
        int rem = e >> 5;
        int xx = rem % PAD, yy = rem / PAD;
        int gy = refl(y0 - RAD + yy, H);
        int gx = refl(x0 - RAD + xx, W);
        sp[e] = proj[((size_t)(b * KD + k) * H + gy) * W + gx];
    }
    __syncthreads();
    int x = x0 + threadIdx.x, y = y0 + threadIdx.y;
    if (x >= W || y >= H) return;
    float cvec[KD];
    const float* ctr = &sp[((threadIdx.y + RAD) * PAD + (threadIdx.x + RAD)) * KD];
    #pragma unroll
    for (int k = 0; k < KD; k++) cvec[k] = ctr[k];
    float temp = expf(rt[0]);
    temp = fminf(fmaxf(temp, 1e-4f), 1e4f);
    float sig = ss[0];
    float vals[D2];
    float m = -1e30f;
    for (int p = 0; p < D2; p++) {
        int i = p / DIA, j = p % DIA;
        const float* nb = &sp[((threadIdx.y + i) * PAD + (threadIdx.x + j)) * KD];
        float d = 0.0f;
        #pragma unroll
        for (int k = 0; k < KD; k++) d += cvec[k] * nb[k];
        d *= temp;
        vals[p] = d;
        m = fmaxf(m, d);
    }
    float sum = 0.0f;
    #pragma unroll
    for (int p = 0; p < D2; p++) { float e2 = expf(vals[p] - m); vals[p] = e2; sum += e2; }
    float inv = 1.0f / sum;
    float s2 = 2.0f * sig * sig;
    float csum = 0.0f;
    #pragma unroll
    for (int p = 0; p < D2; p++) {
        int i = p / DIA, j = p % DIA;
        float dy = -1.0f + (float)i * (1.0f / 3.0f);
        float dx = -1.0f + (float)j * (1.0f / 3.0f);
        float sk = expf(-(dx * dx + dy * dy) / s2);
        float v = vals[p] * inv * sk;
        vals[p] = v;
        csum += v;
    }
    float cinv = 1.0f / fmaxf(csum, 1e-7f);
    #pragma unroll
    for (int p = 0; p < D2; p++)
        ck[((size_t)(b * D2 + p) * H + y) * W + x] = vals[p] * cinv;
}

// ------------- combine fixup MLP: concat(ck,g) 52 -> 49 (gelu) -> 49 -------------
__global__ void fix_kernel(const float* __restrict__ g,
                           const float* __restrict__ fw1, const float* __restrict__ fb1,
                           const float* __restrict__ fw2, const float* __restrict__ fb2,
                           float* __restrict__ ck, int H, int W) {
    int idx = blockIdx.x * 256 + threadIdx.x;
    int n = BB * H * W;
    if (idx >= n) return;
    int x = idx % W;
    int y = (idx / W) % H;
    int b = idx / (W * H);
    float in[D2 + 3];
    for (int p = 0; p < D2; p++) in[p] = ck[((size_t)(b * D2 + p) * H + y) * W + x];
    #pragma unroll
    for (int c = 0; c < 3; c++) in[D2 + c] = g[((size_t)(b * 3 + c) * H + y) * W + x];
    float h1[D2];
    for (int o = 0; o < D2; o++) {
        float a = fb1[o];
        const float* wr = fw1 + o * (D2 + 3);
        #pragma unroll
        for (int i2 = 0; i2 < D2 + 3; i2++) a += wr[i2] * in[i2];
        h1[o] = gelu_exact(a);
    }
    for (int o = 0; o < D2; o++) {
        float a = fb2[o];
        const float* wr = fw2 + o * D2;
        #pragma unroll
        for (int i2 = 0; i2 < D2; i2++) a += wr[i2] * h1[i2];
        ck[((size_t)(b * D2 + o) * H + y) * W + x] = in[o] + 0.1f * a;
    }
}

// ------- bicubic x2 upsample (fused into LDS tile) + 7x7 adaptive conv -------
__global__ void apply_kernel(const float* __restrict__ src, const float* __restrict__ ck,
                             float* __restrict__ out, int h, int w, int H, int W) {
    __shared__ float hr[PAD * PAD];
    int bc = blockIdx.z;
    int b = bc / CC, c = bc % CC;
    const float* sp = src + (size_t)(b * CC + c) * h * w;
    int x0 = blockIdx.x * TS, y0 = blockIdx.y * TS;
    int tid = threadIdx.y * TS + threadIdx.x;
    for (int e = tid; e < PAD * PAD; e += 256) {
        int xx = e % PAD, yy = e / PAD;
        int gy = refl(y0 - RAD + yy, H);
        int gx = refl(x0 - RAD + xx, W);
        hr[e] = bicubic2x(sp, h, w, gy, gx);
    }
    __syncthreads();
    int x = x0 + threadIdx.x, y = y0 + threadIdx.y;
    if (x >= W || y >= H) return;
    const float* ckb = ck + ((size_t)b * D2 * H + y) * W + x;
    size_t plane = (size_t)H * W;
    float acc = 0.0f;
    #pragma unroll
    for (int p = 0; p < D2; p++) {
        int i = p / DIA, j = p % DIA;
        acc += hr[(threadIdx.y + i) * PAD + (threadIdx.x + j)] * ckb[(size_t)p * plane];
    }
    out[((size_t)(b * CC + c) * H + y) * W + x] = acc;
}

// ---------------- f32 -> bf16 pack of the 384x384 weight ----------------
__global__ void pack_w_kernel(const float* __restrict__ Wm, __bf16* __restrict__ Wb) {
    int idx = blockIdx.x * 256 + threadIdx.x;
    if (idx >= CC * CC) return;
    Wb[idx] = (__bf16)Wm[idx];
}

// ---- LDS-tiled transpose+convert: cur [B][C][HW] f32 -> X [B][HW][C] bf16 ----
// Tile: 32 n x 64 c. Reads coalesced along n; writes 16B (8 bf16) per thread.
__global__ void pack_x_kernel(const float* __restrict__ cur, __bf16* __restrict__ Xb,
                              int HW) {
    __shared__ unsigned short lds[32][72];   // 64 c + pad
    int b = blockIdx.z;
    int n0 = blockIdx.x * 32;
    int c0 = blockIdx.y * 64;
    int t = threadIdx.x;
    int tn = t & 31;          // n within tile
    int tg = t >> 5;          // 0..7 -> c group of 8
    const float* base = cur + (size_t)b * CC * HW;
    #pragma unroll
    for (int i = 0; i < 8; i++) {
        int c = tg * 8 + i;
        float v = base[(size_t)(c0 + c) * HW + n0 + tn];
        __bf16 hv = (__bf16)v;
        unsigned short u;
        __builtin_memcpy(&u, &hv, 2);
        lds[tn][c] = u;
    }
    __syncthreads();
    int n = t >> 3;           // 0..31
    int cg = (t & 7) * 8;     // 0..56
    uint4 v4;
    unsigned short* pv = (unsigned short*)&v4;
    #pragma unroll
    for (int i = 0; i < 8; i++) pv[i] = lds[n][cg + i];
    __bf16* dst = Xb + ((size_t)b * HW + n0 + n) * CC + c0 + cg;
    *(uint4*)dst = v4;
}

// load one WMMA 16-half fragment: two aligned 16B runs (K kb..kb+7, kb+16..kb+23)
__device__ __forceinline__ v16bf ldfrag(const __bf16* __restrict__ p) {
    v8bf lo = *(const v8bf*)p;
    v8bf hi = *(const v8bf*)(p + 16);
    v16bf r;
    #pragma unroll
    for (int i = 0; i < 8; i++) { r[i] = lo[i]; r[i + 8] = hi[i]; }
    return r;
}

// --------- final stack conv1x1 as WMMA GEMM: out = 0.1*(W@X + b) + X ---------
// Per wave: 32(M) x 64(N) tile, K over C=384 in steps of 32 -> 8 WMMAs/step.
// Both operands pre-packed bf16 K-contiguous: all fragment loads are b128.
__global__ void stack_gemm_kernel(const __bf16* __restrict__ Xb,   // [B][HW][C]
                                  const __bf16* __restrict__ Wb,   // [C][C]
                                  const float* __restrict__ cur,   // residual [B][C][HW]
                                  const float* __restrict__ bias,
                                  float* __restrict__ out, int HW) {
    int b = blockIdx.z;
    int lane = threadIdx.x & 31;
    int wave = threadIdx.x >> 5;
    int row16 = lane & 15;
    int hi = lane >> 4;                 // K half-group select
    int mbase = blockIdx.x * 32;
    int nbase = (blockIdx.y * 4 + wave) * 64;
    const __bf16* a0p = Wb + (size_t)(mbase + row16) * CC;
    const __bf16* a1p = a0p + (size_t)16 * CC;
    const __bf16* xp[4];
    #pragma unroll
    for (int f = 0; f < 4; f++)
        xp[f] = Xb + ((size_t)b * HW + nbase + f * 16 + row16) * CC;
    v8f acc[2][4] = {};
    for (int k0 = 0; k0 < CC; k0 += 32) {
        int kb = k0 + hi * 8;
        v16bf a0 = ldfrag(a0p + kb);
        v16bf a1 = ldfrag(a1p + kb);
        #pragma unroll
        for (int f = 0; f < 4; f++) {
            __builtin_prefetch((const void*)(xp[f] + kb + 32), 0, 3);
            v16bf bf = ldfrag(xp[f] + kb);
            acc[0][f] = __builtin_amdgcn_wmma_f32_16x16x32_bf16(
                false, a0, false, bf, (short)0, acc[0][f], false, false);
            acc[1][f] = __builtin_amdgcn_wmma_f32_16x16x32_bf16(
                false, a1, false, bf, (short)0, acc[1][f], false, false);
        }
    }
    const float* curb = cur + (size_t)b * CC * HW;
    float* outb = out + (size_t)b * CC * HW;
    #pragma unroll
    for (int mi = 0; mi < 2; mi++) {
        #pragma unroll
        for (int f = 0; f < 4; f++) {
            int n = nbase + f * 16 + row16;
            #pragma unroll
            for (int v = 0; v < 8; v++) {
                int m = mbase + mi * 16 + hi * 8 + v;
                float val = 0.1f * (acc[mi][f][v] + bias[m]) + curb[(size_t)m * HW + n];
                outb[(size_t)m * HW + n] = val;
            }
        }
    }
}

extern "C" void kernel_launch(void* const* d_in, const int* in_sizes, int n_in,
                              void* d_out, int out_size, void* d_ws, size_t ws_size,
                              hipStream_t stream) {
    const float* source        = (const float*)d_in[0];
    const float* guidance      = (const float*)d_in[1];
    const float* range_temp    = (const float*)d_in[2];
    const float* sigma_spatial = (const float*)d_in[3];
    const float* rp_w1 = (const float*)d_in[4];
    const float* rp_b1 = (const float*)d_in[5];
    const float* rp_w2 = (const float*)d_in[6];
    const float* rp_b2 = (const float*)d_in[7];
    const float* fp_w1 = (const float*)d_in[8];
    const float* fp_b1 = (const float*)d_in[9];
    const float* fp_w2 = (const float*)d_in[10];
    const float* fp_b2 = (const float*)d_in[11];
    const float* stack_w = (const float*)d_in[12];
    const float* stack_b = (const float*)d_in[13];

    // workspace layout
    char* wp = (char*)d_ws;
    size_t off = 0;
    auto take = [&](size_t bytes) {
        void* p = wp + off;
        off += (bytes + 255) & ~(size_t)255;
        return p;
    };
    const int HW = 224 * 224;
    const size_t curBytes = (size_t)BB * CC * HW * sizeof(float);
    float*  cur0 = (float*)take(curBytes);
    float*  cur1 = (float*)take(curBytes);
    float*  gbuf = (float*)take((size_t)BB * 3 * HW * sizeof(float));
    float*  proj = (float*)take((size_t)BB * KD * HW * sizeof(float));
    float*  ckb  = (float*)take((size_t)BB * D2 * HW * sizeof(float));
    __bf16* Xb16 = (__bf16*)take((size_t)BB * HW * CC * sizeof(__bf16));
    __bf16* Wb16 = (__bf16*)take((size_t)CC * CC * sizeof(__bf16));
    (void)ws_size; (void)n_in; (void)in_sizes; (void)out_size;

    for (int s = 0; s < 4; s++) {
        int h = 14 << s;
        int H = 2 * h, W = 2 * h;
        int f = 224 / H;
        const float* srcb = (s == 0) ? source : ((s % 2 == 0) ? cur1 : cur0);
        float* dstb = (s % 2 == 0) ? cur0 : cur1;

        int npool = BB * 3 * H * W;
        pool_kernel<<<(npool + 255) / 256, 256, 0, stream>>>(guidance, gbuf, H, W, f);

        int npix = BB * H * W;
        proj_kernel<<<(npix + 255) / 256, 256, 0, stream>>>(
            gbuf, rp_w1 + s * KD * 3, rp_b1 + s * KD,
            rp_w2 + s * KD * KD, rp_b2 + s * KD, proj, H, W);

        dim3 tgrid((W + TS - 1) / TS, (H + TS - 1) / TS, BB);
        ck_kernel<<<tgrid, dim3(TS, TS), 0, stream>>>(
            proj, range_temp + s, sigma_spatial + s, ckb, H, W);

        fix_kernel<<<(npix + 255) / 256, 256, 0, stream>>>(
            gbuf, fp_w1 + s * D2 * (D2 + 3), fp_b1 + s * D2,
            fp_w2 + s * D2 * D2, fp_b2 + s * D2, ckb, H, W);

        dim3 agrid((W + TS - 1) / TS, (H + TS - 1) / TS, BB * CC);
        apply_kernel<<<agrid, dim3(TS, TS), 0, stream>>>(srcb, ckb, dstb, h, h, H, W);
    }

    // pack operands for the WMMA GEMM
    pack_w_kernel<<<(CC * CC + 255) / 256, 256, 0, stream>>>(stack_w, Wb16);
    dim3 pgrid(HW / 32, CC / 64, BB);
    pack_x_kernel<<<pgrid, 256, 0, stream>>>(cur1, Xb16, HW);

    // final stack conv + residual on cur1 (B,384,224,224)
    dim3 ggrid(CC / 32, HW / 256, BB);   // 12 x 196 x 4, 4 waves/block
    stack_gemm_kernel<<<ggrid, 128, 0, stream>>>(Xb16, Wb16, cur1, stack_b,
                                                 (float*)d_out, HW);
}
